// TemporalGraphLearner_67207648248054
// MI455X (gfx1250) — matrix operand order repaired and checked by reading, not compiled, not run on hardware
//
#include <hip/hip_runtime.h>
#include <hip/hip_bf16.h>
#include <cmath>

// ---------------------------------------------------------------------------
// Types for CDNA5 WMMA (wave32) and TDM descriptors
// ---------------------------------------------------------------------------
typedef __attribute__((ext_vector_type(16))) __bf16          v16bf;
typedef __attribute__((ext_vector_type(16))) unsigned short  v16us;
typedef __attribute__((ext_vector_type(8)))  unsigned short  v8us;
typedef __attribute__((ext_vector_type(8)))  float           v8f;
typedef __attribute__((ext_vector_type(4)))  unsigned int    u32x4;
typedef __attribute__((ext_vector_type(8)))  int             i32x8;
typedef __attribute__((ext_vector_type(4)))  int             i32x4;

__device__ __forceinline__ unsigned short f32_to_bf16_rne(float f) {
  unsigned int u = __float_as_uint(f);
  u += 0x7FFFu + ((u >> 16) & 1u);   // round-to-nearest-even
  return (unsigned short)(u >> 16);
}

// ---------------------------------------------------------------------------
// TDM: async 2D bf16 tile load (tile_dim = 32 k  x 128 rows) into LDS.
// D# fields per CDNA5 ISA 8.3/8.4. Rows beyond rows_remaining zero-fill.
// Issue from ONE wave only (TDM ignores EXEC); tracked by TENSORcnt.
// ---------------------------------------------------------------------------
__device__ __forceinline__ void tdm_load_tile_bf16(
    unsigned lds_addr, const unsigned short* gptr,
    int k_remaining, int rows_remaining, int ldk /* elems */) {
  unsigned long long ga = (unsigned long long)(const void*)gptr;
  u32x4 g0;
  g0[0] = 1u;                                    // count=1 (valid), user mode
  g0[1] = lds_addr;                              // LDS byte address
  g0[2] = (unsigned)(ga & 0xFFFFFFFFu);          // global_addr[31:0]
  g0[3] = (unsigned)((ga >> 32) & 0x01FFFFFFu)   // global_addr[56:32]
        | (2u << 30);                            // type=2 ("image")
  i32x8 g1;
  g1[0] = 0x00010000;                            // data_size=1 (2B), mask=0
  g1[1] = (int)(((unsigned)k_remaining & 0xFFFFu) << 16);        // td0 lo
  g1[2] = (int)((((unsigned)k_remaining >> 16) & 0xFFFFu)
        | (((unsigned)rows_remaining & 0xFFFFu) << 16));         // td0 hi|td1 lo
  g1[3] = (int)((((unsigned)rows_remaining >> 16) & 0xFFFFu)
        | (32u << 16));                          // td1 hi | tile_dim0 = 32
  g1[4] = 128;                                   // tile_dim1 = 128, tile_dim2=0
  g1[5] = ldk;                                   // tensor_dim0_stride[31:0]
  g1[6] = 0;                                     // stride hi | dim1_stride lo
  g1[7] = 0;
  i32x4 z4 = {0, 0, 0, 0};
#if __clang_major__ >= 23
  i32x8 z8 = {};
  __builtin_amdgcn_tensor_load_to_lds(g0, g1, z4, z4, z8, 0);
#else
  __builtin_amdgcn_tensor_load_to_lds(g0, g1, z4, z4, 0);
#endif
}

// ---------------------------------------------------------------------------
// Tiled WMMA GEMM:  C[M,Nn] = A[M,K](f32) * Bbf[Nn][K](bf16,k-contig) (+bias)(+=C)
//   Requires K % 32 == 0 (all uses: 128 / 256).
// Block: 256 thr = 8 waves; block tile 64x128; wave tile 16x64 (4 accums).
//   A: staged via VALU f32->bf16 into LDS [64][32] (row-major, 64B rows).
//   B: staged by the Tensor Data Mover, double-buffered, s_wait_tensorcnt(1)
//      overlaps DMA of tile i+1 with WMMA on tile i.
// Fragment reads are 2x ds_load_b128 per operand (per ISA 16-bit layouts).
// ---------------------------------------------------------------------------
template <bool ACC, bool BIAS>
__global__ __launch_bounds__(256) void wmma_gemm_kernel(
    const float* __restrict__ A, int lda,
    const unsigned short* __restrict__ Bbf,    // [Nn][K]
    const float* __restrict__ bias,
    float* __restrict__ C, int ldc,
    int M, int K, int Nn) {
  constexpr int BM = 64, BN = 128, BK = 32;
  constexpr int BTILE = BN * BK;                         // elems per B buffer
  __shared__ alignas(32) unsigned short Ash[BM * BK];    // 4 KB
  __shared__ alignas(64) unsigned short Bsh[2 * BTILE];  // 16 KB, double buffer

  const int tid  = threadIdx.x;
  const int wave = tid >> 5;      // 0..7
  const int lane = tid & 31;
  const int grp  = lane >> 4;     // 0/1 (half-wave)
  const int idx  = lane & 15;
  const int wr   = wave >> 1;     // 0..3 : 16-row tile
  const int cg   = wave & 1;      // 0..1 : 64-col group

  const int rowBase = blockIdx.x * BM;
  const int colBase = blockIdx.y * BN;

  const unsigned BshAddr = (unsigned)(unsigned long long)(const void*)Bsh;
  const unsigned short* Bblk = Bbf + (unsigned long long)colBase * K;
  const int rowsRem = Nn - colBase;

  // A loader geometry
  const int a_row = tid >> 2;                  // 0..63
  const int a_j   = tid & 3;                   // k-octet
  int a_gr = rowBase + a_row;                  // clamped (never stored OOB)
  a_gr = a_gr < M ? a_gr : (M - 1);

  const bool tdmWave = (__builtin_amdgcn_readfirstlane(wave) == 0);

  // prologue: kick off DMA of first B tile into buffer 0
  if (tdmWave)
    tdm_load_tile_bf16(BshAddr, Bblk, K, rowsRem, K);

  v8f acc[4] = {};
  int buf = 0;

  for (int k0 = 0; k0 < K; k0 += BK, buf ^= 1) {
    // ---- stage A tile: 8 consecutive k per thread -> 1 ds_store_b128 ----
    {
      const float* src = A + (long long)a_gr * lda + (k0 + 8 * a_j);
      v8us t;
#pragma unroll
      for (int e = 0; e < 8; ++e) t[e] = f32_to_bf16_rne(src[e]);
      *(v8us*)&Ash[a_row * BK + 8 * a_j] = t;
      __builtin_prefetch(src + BK, 0, 1);  // speculative global_prefetch_b8
    }
    // ---- B pipeline: issue DMA for tile k0+BK, wait for tile k0 ----
    if (tdmWave) {
      if (k0 + BK < K) {
        tdm_load_tile_bf16(BshAddr + (unsigned)((buf ^ 1) * BTILE * 2),
                           Bblk + (k0 + BK), K - (k0 + BK), rowsRem, K);
        __builtin_amdgcn_s_wait_tensorcnt(1);  // tile k0 done (in-order)
      } else {
        __builtin_amdgcn_s_wait_tensorcnt(0);
      }
    }
    __syncthreads();

    // ---- fragments (2x ds_load_b128 each) + 4 WMMAs ----
    const int arow_off = (wr * 16 + idx) * BK;
    v8us alo = *(const v8us*)&Ash[arow_off + 8 * grp];        // K 8g..8g+7
    v8us ahi = *(const v8us*)&Ash[arow_off + 16 + 8 * grp];   // K 16+8g..23+8g
    v16us au = __builtin_shufflevector(alo, ahi, 0, 1, 2, 3, 4, 5, 6, 7,
                                       8, 9, 10, 11, 12, 13, 14, 15);
    v16bf a = __builtin_bit_cast(v16bf, au);
    const unsigned short* Bcur = &Bsh[buf * BTILE];
#pragma unroll
    for (int j = 0; j < 4; ++j) {
      // row (col of C) = cg*64 + j*16 + idx; lane grp g holds K 8g..8g+7, 16+8g..23+8g
      const unsigned short* brow = &Bcur[(cg * 64 + j * 16 + idx) * BK];
      v8us blo = *(const v8us*)&brow[8 * grp];
      v8us bhi = *(const v8us*)&brow[16 + 8 * grp];
      v16us bu = __builtin_shufflevector(blo, bhi, 0, 1, 2, 3, 4, 5, 6, 7,
                                         8, 9, 10, 11, 12, 13, 14, 15);
      v16bf b = __builtin_bit_cast(v16bf, bu);
      acc[j] = __builtin_amdgcn_wmma_f32_16x16x32_bf16(
          /*neg_a=*/false, a, /*neg_b=*/false, b,
          /*c_mod=*/(short)0, acc[j], /*reuse_a=*/false, /*reuse_b=*/false);
    }
    __syncthreads();
  }

  // ---- store: C VGPR v -> row = v + 8*grp (per ISA C/D layout) ----
#pragma unroll
  for (int j = 0; j < 4; ++j) {
    const int col = colBase + cg * 64 + j * 16 + idx;
    if (col < Nn) {
      const float bv = BIAS ? bias[col] : 0.0f;
#pragma unroll
      for (int v = 0; v < 8; ++v) {
        const int row = rowBase + wr * 16 + 8 * grp + v;
        if (row < M) {
          const long long off = (long long)row * ldc + col;
          float r = acc[j][v] + bv;
          if (ACC) r += C[off];
          C[off] = r;
        }
      }
    }
  }
}

// ---------------------------------------------------------------------------
// Weight pre-conversion: out[n*K+k] = bf16( TRANSPOSE ? in[k*Nn+n] : in[n*K+k] )
// Result layout [Nn][K] (k contiguous) feeds the TDM B path.
// ---------------------------------------------------------------------------
template <bool TRANSPOSE>
__global__ void convert_weight_kernel(const float* __restrict__ in,
                                      unsigned short* __restrict__ out,
                                      int K, int Nn) {
  long long i = (long long)blockIdx.x * blockDim.x + threadIdx.x;
  if (i >= (long long)K * Nn) return;
  int n = (int)(i / K), k = (int)(i % K);
  float v = TRANSPOSE ? in[(long long)k * Nn + n] : in[i];
  out[i] = f32_to_bf16_rne(v);
}

// ---------------------------------------------------------------------------
// Elementwise / graph kernels
// ---------------------------------------------------------------------------
__global__ void fill_kernel(float* __restrict__ p, float v, long long n) {
  long long i = (long long)blockIdx.x * blockDim.x + threadIdx.x;
  if (i < n) p[i] = v;
}

__global__ void count_deg_kernel(const long long* __restrict__ ei,
                                 float* __restrict__ deg, int E) {
  int e = blockIdx.x * blockDim.x + threadIdx.x;
  if (e < E) atomicAdd(&deg[ei[(long long)E + e]], 1.0f);
}

__global__ void rsqrt_kernel(float* __restrict__ p, int n) {
  int i = blockIdx.x * blockDim.x + threadIdx.x;
  if (i < n) p[i] = rsqrtf(p[i]);   // deg >= 1 (self-loop) so always valid
}

// agg[dst] += dinv[src]*dinv[dst] * xw[src]   (one block per edge, H threads)
__global__ void aggregate_kernel(const long long* __restrict__ ei,
                                 const float* __restrict__ dinv,
                                 const float* __restrict__ xw,
                                 float* __restrict__ agg, int E, int H) {
  int e = blockIdx.x;
  long long s = ei[e];
  long long d = ei[(long long)E + e];
  float nrm = dinv[s] * dinv[d];
  int c = threadIdx.x;
  atomicAdd(&agg[d * (long long)H + c], nrm * xw[s * (long long)H + c]);
}

// out = relu(agg + dinv[n]^2 * xw[n] (self-loop) + bias)
__global__ void relu_combine_kernel(const float* __restrict__ agg,
                                    const float* __restrict__ xw,
                                    const float* __restrict__ dinv,
                                    const float* __restrict__ bias,
                                    float* __restrict__ out, int H) {
  long long n = blockIdx.x;
  int c = threadIdx.x;
  float di = dinv[n];
  float v = agg[n * H + c] + di * di * xw[n * H + c] + bias[c];
  out[n * H + c] = v > 0.0f ? v : 0.0f;
}

// LSTM cell pointwise update; gates layout [N, 4H] = [i|f|g|o]
__global__ void lstm_pointwise_kernel(const float* __restrict__ gates,
                                      const float* __restrict__ b_ih,
                                      const float* __restrict__ b_hh,
                                      float* __restrict__ hbuf,
                                      float* __restrict__ cbuf,
                                      int N, int H) {
  long long t = (long long)blockIdx.x * blockDim.x + threadIdx.x;
  if (t >= (long long)N * H) return;
  int n = (int)(t / H), c = (int)(t % H);
  const float* g4 = gates + (long long)n * 4 * H;
  float gi = g4[c]         + b_ih[c]         + b_hh[c];
  float gf = g4[H + c]     + b_ih[H + c]     + b_hh[H + c];
  float gg = g4[2 * H + c] + b_ih[2 * H + c] + b_hh[2 * H + c];
  float go = g4[3 * H + c] + b_ih[3 * H + c] + b_hh[3 * H + c];
  float i_ = 1.0f / (1.0f + __expf(-gi));
  float f_ = 1.0f / (1.0f + __expf(-gf));
  float o_ = 1.0f / (1.0f + __expf(-go));
  float cc = f_ * cbuf[t] + i_ * tanhf(gg);
  cbuf[t] = cc;
  hbuf[t] = o_ * tanhf(cc);
}

// ---------------------------------------------------------------------------
// Host orchestration
// ---------------------------------------------------------------------------
extern "C" void kernel_launch(void* const* d_in, const int* in_sizes, int n_in,
                              void* d_out, int out_size, void* d_ws, size_t ws_size,
                              hipStream_t stream) {
  (void)n_in; (void)out_size; (void)ws_size;
  const long long* ei = (const long long*)d_in[0];
  const float* x_seq  = (const float*)d_in[1];
  const float* W_g0   = (const float*)d_in[2];
  const float* b_g0   = (const float*)d_in[3];
  const float* W_g1   = (const float*)d_in[4];
  const float* b_g1   = (const float*)d_in[5];
  const float* W_ih   = (const float*)d_in[6];
  const float* W_hh   = (const float*)d_in[7];
  const float* b_ih   = (const float*)d_in[8];
  const float* b_hh   = (const float*)d_in[9];
  const float* W_fc   = (const float*)d_in[10];
  const float* b_fc   = (const float*)d_in[11];

  const int H = in_sizes[3];               // 256
  const int C = in_sizes[2] / H;           // 128
  const int T = 12;
  const int N = in_sizes[1] / (T * C);     // 20000
  const int E = in_sizes[0] / 2;           // 640000
  const int O = in_sizes[11];              // 32

  // carve scratch
  char* ws = (char*)d_ws;
  auto carve = [&](size_t bytes) -> char* {
    char* p = ws;
    ws += (bytes + 255) & ~(size_t)255;
    return p;
  };
  float* dinv  = (float*)carve(sizeof(float) * (size_t)N);
  float* bufA  = (float*)carve(sizeof(float) * (size_t)N * H);   // xw
  float* bufB  = (float*)carve(sizeof(float) * (size_t)N * H);   // agg
  float* bufG  = (float*)carve(sizeof(float) * (size_t)N * H);   // layer out / g_t
  float* hbuf  = (float*)carve(sizeof(float) * (size_t)N * H);
  float* cbuf  = (float*)carve(sizeof(float) * (size_t)N * H);
  float* gates = (float*)carve(sizeof(float) * (size_t)N * 4 * H);
  unsigned short* Wg0t = (unsigned short*)carve(2ull * C * H);      // [H][C]
  unsigned short* Wg1t = (unsigned short*)carve(2ull * H * H);      // [H][H]
  unsigned short* Wiht = (unsigned short*)carve(2ull * 4 * H * H);  // [4H][H]
  unsigned short* Whht = (unsigned short*)carve(2ull * 4 * H * H);  // [4H][H]
  unsigned short* Wfct = (unsigned short*)carve(2ull * H * O);      // [O][H]

  const dim3 blk(256);
  const long long NH = (long long)N * H;
  const unsigned nhBlocks = (unsigned)((NH + 255) / 256);

  // weight pre-conversion to bf16 [Nn][K]
  convert_weight_kernel<true><<<dim3((C * H + 255) / 256), blk, 0, stream>>>(W_g0, Wg0t, C, H);
  convert_weight_kernel<true><<<dim3((H * H + 255) / 256), blk, 0, stream>>>(W_g1, Wg1t, H, H);
  convert_weight_kernel<false><<<dim3((4 * H * H + 255) / 256), blk, 0, stream>>>(W_ih, Wiht, H, 4 * H);
  convert_weight_kernel<false><<<dim3((4 * H * H + 255) / 256), blk, 0, stream>>>(W_hh, Whht, H, 4 * H);
  convert_weight_kernel<true><<<dim3((H * O + 255) / 256), blk, 0, stream>>>(W_fc, Wfct, H, O);

  // degree / normalization
  fill_kernel<<<dim3((N + 255) / 256), blk, 0, stream>>>(dinv, 1.0f, N);
  count_deg_kernel<<<dim3((E + 255) / 256), blk, 0, stream>>>(ei, dinv, E);
  rsqrt_kernel<<<dim3((N + 255) / 256), blk, 0, stream>>>(dinv, N);

  // zero LSTM state
  fill_kernel<<<dim3(nhBlocks), blk, 0, stream>>>(hbuf, 0.0f, NH);
  fill_kernel<<<dim3(nhBlocks), blk, 0, stream>>>(cbuf, 0.0f, NH);

  const dim3 gridH((N + 63) / 64, (H + 127) / 128);
  const dim3 gridGates((N + 63) / 64, (4 * H + 127) / 128);
  const dim3 gridOut((N + 63) / 64, (O + 127) / 128);

  for (int t = 0; t < T; ++t) {
    // ---- GCN layer 0: xw = x_t @ W_g0 ----
    wmma_gemm_kernel<false, false><<<gridH, blk, 0, stream>>>(
        x_seq + (size_t)t * C, T * C, Wg0t, nullptr, bufA, H, N, C, H);
    fill_kernel<<<dim3(nhBlocks), blk, 0, stream>>>(bufB, 0.0f, NH);
    aggregate_kernel<<<dim3(E), dim3(H), 0, stream>>>(ei, dinv, bufA, bufB, E, H);
    relu_combine_kernel<<<dim3(N), dim3(H), 0, stream>>>(bufB, bufA, dinv, b_g0, bufG, H);

    // ---- GCN layer 1: xw = g @ W_g1 ----
    wmma_gemm_kernel<false, false><<<gridH, blk, 0, stream>>>(
        bufG, H, Wg1t, nullptr, bufA, H, N, H, H);
    fill_kernel<<<dim3(nhBlocks), blk, 0, stream>>>(bufB, 0.0f, NH);
    aggregate_kernel<<<dim3(E), dim3(H), 0, stream>>>(ei, dinv, bufA, bufB, E, H);
    relu_combine_kernel<<<dim3(N), dim3(H), 0, stream>>>(bufB, bufA, dinv, b_g1, bufG, H);

    // ---- LSTM gates: gates = g_t @ W_ih^T + h @ W_hh^T ----
    wmma_gemm_kernel<false, false><<<gridGates, blk, 0, stream>>>(
        bufG, H, Wiht, nullptr, gates, 4 * H, N, H, 4 * H);
    wmma_gemm_kernel<true, false><<<gridGates, blk, 0, stream>>>(
        hbuf, H, Whht, nullptr, gates, 4 * H, N, H, 4 * H);
    lstm_pointwise_kernel<<<dim3(nhBlocks), blk, 0, stream>>>(
        gates, b_ih, b_hh, hbuf, cbuf, N, H);
  }

  // ---- final FC: out = h_last @ W_fc + b_fc ----
  wmma_gemm_kernel<false, true><<<gridOut, blk, 0, stream>>>(
      hbuf, H, Wfct, b_fc, (float*)d_out, O, N, H, O);
}
